// Model_2619930051388
// MI455X (gfx1250) — matrix operand order repaired and checked by reading, not compiled, not run on hardware
//
#include <hip/hip_runtime.h>
#include <stdint.h>

typedef __attribute__((ext_vector_type(16))) __bf16 v16bf;
typedef __attribute__((ext_vector_type(8)))  float  v8f;

#define B_    2
#define S_    2048
#define HQ    15
#define HKV   5
#define GRP   3          // Hq / Hkv
#define D_    64
#define SCALE 0.125f     // D^-0.5

#define WAVES 4          // waves per block
#define BM    (WAVES*16) // 64 query rows per block
#define BK    32         // keys per iteration

#define KRS 68           // raw K/V row stride (f32 elems), 16B-granular pad
#define VS  40           // sVT row stride (bf16 elems), padded
#define PS  36           // sP  row stride (bf16 elems), padded

// Async DMA-style copy: 16 bytes global -> LDS, tracked by ASYNCcnt.
__device__ __forceinline__ void async_copy_b128(const float* gsrc, const float* ldst) {
  const uint32_t laddr = (uint32_t)(uintptr_t)ldst;   // low 32 bits = LDS offset
  asm volatile("global_load_async_to_lds_b128 %0, %1, off"
               :: "v"(laddr), "v"(gsrc) : "memory");
}
__device__ __forceinline__ void wait_asynccnt0() {
  asm volatile("s_wait_asynccnt 0" ::: "memory");
}

__global__ __launch_bounds__(128)
void gqa_flash_fwd(const uint8_t* __restrict__ mask,
                   const float*   __restrict__ q,
                   const float*   __restrict__ k,
                   const float*   __restrict__ v,
                   float*         __restrict__ out) {
  __shared__ float  sKraw[2][BK * KRS];   // raw f32 K tile  [key][d], double-buffered
  __shared__ float  sVraw[2][BK * KRS];   // raw f32 V tile  [key][d], double-buffered
  __shared__ __bf16 sVT[D_ * VS];         // bf16 V tile [d][key] (transposed)
  __shared__ __bf16 sP [WAVES * 16 * PS]; // per-wave P transpose buffer

  const int mgrp = blockIdx.x % (S_ / BM);
  const int bh   = blockIdx.x / (S_ / BM);
  const int b    = bh / HQ;
  const int hq   = bh % HQ;
  const int hkv  = hq / GRP;

  const int tid  = threadIdx.x;
  const int wave = tid >> 5;
  const int lane = tid & 31;
  const int l16  = lane & 15;
  const int hi   = (lane >> 4) & 1;

  const int qbase = mgrp * BM + wave * 16;

  // Per-thread staging assignment: key row + 16-float d-chunk.
  const int skey  = tid >> 2;          // 0..31
  const int sdblk = (tid & 3) * 16;    // 0,16,32,48
  const size_t kvRowBase = ((size_t)b * S_) * HKV + hkv;

  // ---- Q A-fragments (16x32 bf16 each; contiguous k = 16*hi + s order), kept in VGPRs ----
  const int qrowA = qbase + l16;
  const float* qp = q + (((size_t)b * S_ + qrowA) * HQ + hq) * D_;
  const int abase = hi ? 16 : 0;
  v16bf qa0, qa1;
#pragma unroll
  for (int i = 0; i < 8; ++i) {
    qa0[i]     = (__bf16)qp[abase + i];
    qa0[8 + i] = (__bf16)qp[abase + 8 + i];
    qa1[i]     = (__bf16)qp[32 + abase + i];
    qa1[8 + i] = (__bf16)qp[32 + abase + 8 + i];
  }

  v8f oacc[4];
#pragma unroll
  for (int t = 0; t < 4; ++t) oacc[t] = (v8f){};
  float mrow[8], lrow[8];
#pragma unroll
  for (int r = 0; r < 8; ++r) { mrow[r] = -3.0e38f; lrow[r] = 0.0f; }

  const uint8_t* mbase = mask + (size_t)b * S_ * S_;

  // ---- prologue: async-stage tile 0 into buffer 0 ----
  {
    const float* kp = k + ((kvRowBase + (size_t)skey * HKV)) * D_ + sdblk;
    const float* vp = v + ((kvRowBase + (size_t)skey * HKV)) * D_ + sdblk;
    float* kl = &sKraw[0][skey * KRS + sdblk];
    float* vl = &sVraw[0][skey * KRS + sdblk];
#pragma unroll
    for (int j = 0; j < 4; ++j) {
      async_copy_b128(kp + j * 4, kl + j * 4);
      async_copy_b128(vp + j * 4, vl + j * 4);
    }
  }

  int cur = 0;
  for (int key0 = 0; key0 < S_; key0 += BK, cur ^= 1) {
    // ---- consume async tile: wait own ASYNCcnt, then make visible block-wide ----
    wait_asynccnt0();
    __syncthreads();

    // ---- software pipeline: kick off async copies for the NEXT tile ----
    if (key0 + BK < S_) {
      const int nxt = cur ^ 1;
      const float* kp = k + ((kvRowBase + (size_t)(key0 + BK + skey) * HKV)) * D_ + sdblk;
      const float* vp = v + ((kvRowBase + (size_t)(key0 + BK + skey) * HKV)) * D_ + sdblk;
      float* kl = &sKraw[nxt][skey * KRS + sdblk];
      float* vl = &sVraw[nxt][skey * KRS + sdblk];
#pragma unroll
      for (int j = 0; j < 4; ++j) {
        async_copy_b128(kp + j * 4, kl + j * 4);
        async_copy_b128(vp + j * 4, vl + j * 4);
      }
    }

    // ---- V: raw f32 -> transposed bf16 for contiguous B-fragment reads ----
    {
      const float* vr = &sVraw[cur][skey * KRS + sdblk];
#pragma unroll
      for (int i = 0; i < 16; ++i)
        sVT[(sdblk + i) * VS + skey] = (__bf16)vr[i];
    }
    __syncthreads();

    // ---- scores: S = Q * K^T, two 16-key tiles, contraction over d=64 ----
    v8f s0 = (v8f){}, s1 = (v8f){};
    {
      const float* krow0 = &sKraw[cur][l16 * KRS];
      const float* krow1 = &sKraw[cur][(16 + l16) * KRS];
      const int dsel = hi ? 16 : 0;    // B layout: lo lanes K 0..15, hi lanes K 16..31
#pragma unroll
      for (int c = 0; c < 2; ++c) {
        const int dbase = c * 32 + dsel;
        v16bf kb0, kb1;
#pragma unroll
        for (int i = 0; i < 16; ++i) {
          kb0[i] = (__bf16)krow0[dbase + i];
          kb1[i] = (__bf16)krow1[dbase + i];
        }
        const v16bf qa = c ? qa1 : qa0;
        s0 = __builtin_amdgcn_wmma_f32_16x16x32_bf16(false, qa, false, kb0,
                                                     (short)0, s0, false, false);
        s1 = __builtin_amdgcn_wmma_f32_16x16x32_bf16(false, qa, false, kb1,
                                                     (short)0, s1, false, false);
      }
    }

    // ---- mask + scale + online softmax (rows in VGPR idx, cols in lanes) ----
    const int col0 = key0 + l16;
    float p0[8], p1[8], cf[8];
#pragma unroll
    for (int r = 0; r < 8; ++r) {
      const int qr = qbase + r + 8 * hi;
      const uint8_t m0 = mbase[(size_t)qr * S_ + col0];
      const uint8_t m1 = mbase[(size_t)qr * S_ + col0 + 16];
      float e0 = m0 ? s0[r] * SCALE : -1.0e30f;
      float e1 = m1 ? s1[r] * SCALE : -1.0e30f;
      float tmax = fmaxf(e0, e1);
#pragma unroll
      for (int off = 1; off < 16; off <<= 1)
        tmax = fmaxf(tmax, __shfl_xor(tmax, off, 32));
      const float mn = fmaxf(mrow[r], tmax);
      const float c  = __expf(mrow[r] - mn);
      e0 = __expf(e0 - mn);
      e1 = __expf(e1 - mn);
      float ts = e0 + e1;
#pragma unroll
      for (int off = 1; off < 16; off <<= 1)
        ts += __shfl_xor(ts, off, 32);
      lrow[r] = lrow[r] * c + ts;
      mrow[r] = mn;
      p0[r] = e0; p1[r] = e1; cf[r] = c;
    }

    // rescale running O by exp(m_old - m_new)
#pragma unroll
    for (int t = 0; t < 4; ++t)
#pragma unroll
      for (int r = 0; r < 8; ++r) oacc[t][r] *= cf[r];

    // ---- P: C-layout -> LDS -> A-layout fragment ----
    __bf16* pw = sP + wave * 16 * PS;
#pragma unroll
    for (int r = 0; r < 8; ++r) {
      const int prow = r + 8 * hi;
      pw[prow * PS + l16]      = (__bf16)p0[r];
      pw[prow * PS + l16 + 16] = (__bf16)p1[r];
    }
    v16bf pa;
    {
      const int pb = hi ? 16 : 0;
#pragma unroll
      for (int i = 0; i < 8; ++i) {
        pa[i]     = pw[l16 * PS + pb + i];
        pa[8 + i] = pw[l16 * PS + pb + 8 + i];
      }
    }

    // ---- O += P(16x32) * V(32x64): four 16-col output tiles ----
    {
      const int kb = hi ? 16 : 0;
#pragma unroll
      for (int t = 0; t < 4; ++t) {
        v16bf vb;
#pragma unroll
        for (int i = 0; i < 16; ++i)
          vb[i] = sVT[(t * 16 + l16) * VS + kb + i];
        oacc[t] = __builtin_amdgcn_wmma_f32_16x16x32_bf16(false, pa, false, vb,
                                                          (short)0, oacc[t], false, false);
      }
    }
    __syncthreads();   // protect sVT / raw buffers before next iteration's writes
  }

  // ---- epilogue: divide by softmax denominator, store [B, S, Hq*D] ----
#pragma unroll
  for (int r = 0; r < 8; ++r) {
    const int qr = qbase + r + 8 * hi;
    const float inv = 1.0f / lrow[r];
    float* orow = out + ((size_t)b * S_ + qr) * (HQ * D_) + hq * D_ + l16;
    orow[0]  = oacc[0][r] * inv;
    orow[16] = oacc[1][r] * inv;
    orow[32] = oacc[2][r] * inv;
    orow[48] = oacc[3][r] * inv;
  }
}

extern "C" void kernel_launch(void* const* d_in, const int* in_sizes, int n_in,
                              void* d_out, int out_size, void* d_ws, size_t ws_size,
                              hipStream_t stream) {
  const uint8_t* mask = (const uint8_t*)d_in[0];
  const float*   q    = (const float*)d_in[1];
  const float*   k    = (const float*)d_in[2];
  const float*   v    = (const float*)d_in[3];
  float*         out  = (float*)d_out;

  dim3 grid(B_ * HQ * (S_ / BM));   // 2*15*32 = 960 blocks
  dim3 block(WAVES * 32);           // 128 threads = 4 wave32
  gqa_flash_fwd<<<grid, block, 0, stream>>>(mask, q, k, v, out);
}
